// LNLE_26328149524722
// MI455X (gfx1250) — matrix operand order repaired and checked by reading, compiled, not run on hardware
//
#include <hip/hip_runtime.h>
#include <hip/hip_bf16.h>

typedef __attribute__((ext_vector_type(16))) _Float16 v16h;
typedef __attribute__((ext_vector_type(8)))  _Float16 v8h;
typedef __attribute__((ext_vector_type(8)))  float    v8f;

#define TM 128
#define TN 128
#define TK 32

// ---------------- order-preserving float <-> uint encode for atomicMax ----
__device__ __forceinline__ unsigned fenc(float f) {
    unsigned u = __float_as_uint(f);
    return (u & 0x80000000u) ? ~u : (u | 0x80000000u);
}
__device__ __forceinline__ float fdec(unsigned u) {
    unsigned v = (u & 0x80000000u) ? (u & 0x7fffffffu) : ~u;
    return __uint_as_float(v);
}
__device__ __forceinline__ float lrelu(float x) { return x > 0.f ? x : 0.2f * x; }

// Async global -> LDS copy, 16 bytes per enabled lane (CDNA5 VFLAT async op).
// lds_u32 = DS byte offset (low 32 bits of generic shared address),
// gp      = 64-bit global address. Tracked with ASYNCcnt.
__device__ __forceinline__ void async_ld_b128(unsigned lds_u32, const void* gp) {
    asm volatile("global_load_async_to_lds_b128 %0, %1, off"
                 :: "v"(lds_u32), "v"(gp)
                 : "memory");
}
__device__ __forceinline__ void wait_async0() {
    asm volatile("s_wait_asynccnt 0" ::: "memory");
}

// ---------------- WMMA GEMM: C[M x ldc] = A[M x K] * Bt[ldc x K]^T --------
// A row-major f16, Bt is B transposed (Ncols x K) f16.
// 256 threads = 8 waves; block tile 128x128; wave tile 64x32 (4x2 wmma tiles).
// Double-buffered LDS, async-DMA staging software-pipelined against WMMA.
__global__ __launch_bounds__(256)
void gemm_f16_wmma(const _Float16* __restrict__ A,
                   const _Float16* __restrict__ Bt,
                   const float* __restrict__ bias,   // length ldc or nullptr
                   float* __restrict__ Cf32,         // nullable
                   _Float16* __restrict__ Cf16,      // nullable
                   int K, int ldc, int act)
{
    __shared__ _Float16 As[2][TM * TK];
    __shared__ _Float16 Bs[2][TN * TK];

    const int tid  = threadIdx.x;
    const int lane = tid & 31;
    const int w    = tid >> 5;
    const int m0   = blockIdx.x * TM;
    const int n0   = blockIdx.y * TN;
    const int rm   = (w & 1) * 64;   // wave row offset in tile
    const int cn   = (w >> 1) * 32;  // wave col offset in tile
    const int lm   = lane & 15;
    const int o1   = (lane < 16) ? 0 : 8;    // A half-offset (K pairs)
    const int klo  = (lane < 16) ? 0 : 8;    // B half-offset (K pairs)
    const int rofs = (lane < 16) ? 0 : 8;    // C row offset

    const v8f vzero = {0.f,0.f,0.f,0.f,0.f,0.f,0.f,0.f};
    v8f acc[4][2];
#pragma unroll
    for (int i = 0; i < 4; ++i)
#pragma unroll
        for (int j = 0; j < 2; ++j) acc[i][j] = vzero;

    union HV { v16h v; v8h h[2]; };

    // Per-thread staging assignment: 512 16B vectors per 128x32 tile,
    // 2 per thread; issued as async DMA (no VGPR bounce, no loadcnt stall).
    auto stage = [&](int kt, int buf) {
        const _Float16* Ag = A  + m0 * K + kt * TK;
        const _Float16* Bg = Bt + n0 * K + kt * TK;
#pragma unroll
        for (int r = 0; r < 2; ++r) {
            int vec = tid + r * 256;       // 0..511
            int row = vec >> 2;            // 0..127
            int cb  = (vec & 3) * 8;       // 0,8,16,24
            unsigned la = (unsigned)(size_t)&As[buf][row * TK + cb];
            unsigned lb = (unsigned)(size_t)&Bs[buf][row * TK + cb];
            async_ld_b128(la, (const void*)(Ag + row * K + cb));
            async_ld_b128(lb, (const void*)(Bg + row * K + cb));
        }
    };

    const int numK = K / TK;
    stage(0, 0);

    for (int kt = 0; kt < numK; ++kt) {
        const int cur = kt & 1;
        wait_async0();        // own wave's async DMA done
        __syncthreads();      // all waves' DMA for chunk kt visible in LDS

        if (kt + 1 < numK) stage(kt + 1, cur ^ 1);  // overlap DMA with WMMA

        HV a[4], b[2];
#pragma unroll
        for (int i = 0; i < 4; ++i) {
            int row = rm + i * 16 + lm;
            a[i].h[0] = *(const v8h*)&As[cur][row * TK + o1];
            a[i].h[1] = *(const v8h*)&As[cur][row * TK + o1 + 16];
        }
#pragma unroll
        for (int j = 0; j < 2; ++j) {
            int col = cn + j * 16 + lm;
            b[j].h[0] = *(const v8h*)&Bs[cur][col * TK + klo];
            b[j].h[1] = *(const v8h*)&Bs[cur][col * TK + klo + 16];
        }
#pragma unroll
        for (int i = 0; i < 4; ++i)
#pragma unroll
            for (int j = 0; j < 2; ++j)
                acc[i][j] = __builtin_amdgcn_wmma_f32_16x16x32_f16(
                    false, a[i].v, false, b[j].v,
                    (short)0, acc[i][j], false, false);
    }

    // epilogue per documented 16x16 f32 C layout
#pragma unroll
    for (int i = 0; i < 4; ++i) {
#pragma unroll
        for (int j = 0; j < 2; ++j) {
            int col = n0 + cn + j * 16 + lm;
            float bv = bias ? bias[col] : 0.f;
#pragma unroll
            for (int v = 0; v < 8; ++v) {
                int row = m0 + rm + i * 16 + rofs + v;
                float val = acc[i][j][v] + bv;
                if (act) val = val > 0.f ? val : 0.f;
                if (Cf32) Cf32[row * ldc + col] = val;
                if (Cf16) Cf16[row * ldc + col] = (_Float16)val;
            }
        }
    }
}

// ---------------- conversion / transpose helpers --------------------------
__global__ void k_f32_to_f16_pad_rows(const float* __restrict__ X,
                                      _Float16* __restrict__ Y,
                                      int n_rows, int cols, int total)
{
    int idx = blockIdx.x * blockDim.x + threadIdx.x;
    if (idx >= total) return;
    int row = idx / cols;
    Y[idx] = (row < n_rows) ? (_Float16)X[idx] : (_Float16)0.f;
}

// W[K x Ncols] f32 -> Wt[Npad x Kpad] f16 (transposed, zero-padded)
__global__ void k_transpose_pad(const float* __restrict__ W,
                                _Float16* __restrict__ Wt,
                                int K, int Ncols, int Kpad, int Npad)
{
    int idx = blockIdx.x * blockDim.x + threadIdx.x;
    int total = Npad * Kpad;
    if (idx >= total) return;
    int o = idx / Kpad;
    int k = idx % Kpad;
    float v = (o < Ncols && k < K) ? W[k * Ncols + o] : 0.f;
    Wt[idx] = (_Float16)v;
}

// ---------------- GAT building blocks -------------------------------------
// hg1[n][o] = sum_{k<16} (h3[n*128+k] + b3[k]) * Wg1[k*40+o]
__global__ void k_gat1_lin(const float* __restrict__ h3,
                           const float* __restrict__ b3,
                           const float* __restrict__ Wg1,
                           float* __restrict__ hg1, int N)
{
    int idx = blockIdx.x * blockDim.x + threadIdx.x;
    if (idx >= N * 40) return;
    int n = idx / 40, o = idx % 40;
    float acc = 0.f;
#pragma unroll
    for (int k = 0; k < 16; ++k)
        acc += (h3[n * 128 + k] + b3[k]) * Wg1[k * 40 + o];
    hg1[idx] = acc;
}

// a_s[n][h] = sum_c hfeat[(n*H+h)*C + c] * att_s[h*C+c]  (same for a_d)
__global__ void k_att_node(const float* __restrict__ hfeat, int C, int H,
                           const float* __restrict__ att_s,
                           const float* __restrict__ att_d,
                           float* __restrict__ a_s, float* __restrict__ a_d,
                           int N)
{
    int idx = blockIdx.x * blockDim.x + threadIdx.x;
    if (idx >= N * H) return;
    int h = idx % H;
    float s = 0.f, d = 0.f;
    for (int c = 0; c < C; ++c) {
        float v = hfeat[idx * C + c];
        s += v * att_s[h * C + c];
        d += v * att_d[h * C + c];
    }
    a_s[idx] = s;
    a_d[idx] = d;
}

__global__ void k_edge_max(const int* __restrict__ sE, const int* __restrict__ tE,
                           const float* __restrict__ a_s, const float* __restrict__ a_d,
                           unsigned* __restrict__ mx, int Edges, int H)
{
    int idx = blockIdx.x * blockDim.x + threadIdx.x;
    if (idx >= Edges * H) return;
    int e = idx / H, h = idx % H;
    int s = sE[e], t = tE[e];
    float l = lrelu(a_s[s * H + h] + a_d[t * H + h]);
    atomicMax(&mx[t * H + h], fenc(l));
}

__global__ void k_edge_expsum(const int* __restrict__ sE, const int* __restrict__ tE,
                              const float* __restrict__ a_s, const float* __restrict__ a_d,
                              const unsigned* __restrict__ mx,
                              float* __restrict__ ex, float* __restrict__ den,
                              int Edges, int H)
{
    int idx = blockIdx.x * blockDim.x + threadIdx.x;
    if (idx >= Edges * H) return;
    int e = idx / H, h = idx % H;
    int s = sE[e], t = tE[e];
    float l = lrelu(a_s[s * H + h] + a_d[t * H + h]);
    float m = fdec(mx[t * H + h]);
    float v = __expf(l - m);
    ex[idx] = v;
    atomicAdd(&den[t * H + h], v);
}

// GAT1 aggregation: per (edge, head), 10 channels
__global__ void k_aggregate1(const int* __restrict__ sE, const int* __restrict__ tE,
                             const float* __restrict__ ex, const float* __restrict__ den,
                             const float* __restrict__ hg1,
                             float* __restrict__ out1, int Edges)
{
    int idx = blockIdx.x * blockDim.x + threadIdx.x;
    if (idx >= Edges * 4) return;
    int e = idx >> 2, h = idx & 3;
    int s = sE[e], t = tE[e];
    float alpha = ex[idx] / (den[t * 4 + h] + 1e-16f);
#pragma unroll
    for (int c = 0; c < 10; ++c)
        atomicAdd(&out1[t * 40 + h * 10 + c], hg1[s * 40 + h * 10 + c] * alpha);
}

// relu(out1 + bg1) -> f16 padded [Mpad x 64]
__global__ void k_relu_bias_to_f16pad(const float* __restrict__ out1,
                                      const float* __restrict__ bg1,
                                      _Float16* __restrict__ Y,
                                      int N, int total)
{
    int idx = blockIdx.x * blockDim.x + threadIdx.x;
    if (idx >= total) return;
    int row = idx >> 6, c = idx & 63;
    float v = 0.f;
    if (row < N && c < 40) {
        v = out1[row * 40 + c] + bg1[c];
        v = v > 0.f ? v : 0.f;
    }
    Y[idx] = (_Float16)v;
}

// GAT2 aggregation: one atomicAdd per (edge, channel)
__global__ void k_aggregate2(const int* __restrict__ sE, const int* __restrict__ tE,
                             const float* __restrict__ ex2, const float* __restrict__ den2,
                             const float* __restrict__ hg2,
                             float* __restrict__ outAcc, int total)
{
    int idx = blockIdx.x * blockDim.x + threadIdx.x;
    if (idx >= total) return;
    int e = idx >> 7, c = idx & 127;
    int s = sE[e], t = tE[e];
    float alpha = ex2[e] / (den2[t] + 1e-16f);
    atomicAdd(&outAcc[t * 128 + c], hg2[s * 128 + c] * alpha);
}

__global__ void k_final_sigmoid(float* __restrict__ out,
                                const float* __restrict__ bg2, int total)
{
    int idx = blockIdx.x * blockDim.x + threadIdx.x;
    if (idx >= total) return;
    int c = idx & 127;
    float z = out[idx] + bg2[c];
    out[idx] = 256.f / (1.f + __expf(-z));
}

// --------------------------------------------------------------------------
extern "C" void kernel_launch(void* const* d_in, const int* in_sizes, int n_in,
                              void* d_out, int out_size, void* d_ws, size_t ws_size,
                              hipStream_t stream) {
    const int N = 100000;
    const int E = N * 9;           // 900000
    const int Mpad = 100096;       // 782 * 128

    const float* x   = (const float*)d_in[0];
    const int*   ei  = (const int*)d_in[1];
    const float* W1  = (const float*)d_in[2];
    const float* b1  = (const float*)d_in[3];
    const float* W2  = (const float*)d_in[4];
    const float* b2  = (const float*)d_in[5];
    const float* W3  = (const float*)d_in[6];
    const float* b3  = (const float*)d_in[7];
    const float* Wg1 = (const float*)d_in[8];
    const float* as1 = (const float*)d_in[9];
    const float* ad1 = (const float*)d_in[10];
    const float* bg1 = (const float*)d_in[11];
    const float* Wg2 = (const float*)d_in[12];
    const float* as2 = (const float*)d_in[13];
    const float* ad2 = (const float*)d_in[14];
    const float* bg2 = (const float*)d_in[15];
    const int* sE = ei;            // edge_index row 0 (source of messages)
    const int* tE = ei + E;        // edge_index row 1 (softmax segment / target)
    float* out = (float*)d_out;

    // ---- workspace bump allocator ----
    char* base = (char*)d_ws;
    size_t off = 0;
    auto alloc = [&](size_t bytes) -> void* {
        off = (off + 255) & ~(size_t)255;
        void* r = base + off;
        off += bytes;
        return r;
    };
    _Float16* bufA  = (_Float16*)alloc((size_t)Mpad * 256 * 2);
    _Float16* bufB  = (_Float16*)alloc((size_t)Mpad * 256 * 2);
    float*    fbuf  = (float*)   alloc((size_t)Mpad * 128 * 4); // h3, later hg2
    _Float16* W1t   = (_Float16*)alloc((size_t)256 * 128 * 2);
    _Float16* W2t   = (_Float16*)alloc((size_t)256 * 256 * 2);
    _Float16* W3t   = (_Float16*)alloc((size_t)128 * 256 * 2);
    _Float16* Wg2t  = (_Float16*)alloc((size_t)128 * 64 * 2);
    _Float16* h1g16 = (_Float16*)alloc((size_t)Mpad * 64 * 2);
    float*    hg1   = (float*)   alloc((size_t)N * 40 * 4);
    float*    as1b  = (float*)   alloc((size_t)N * 4 * 4);
    float*    ad1b  = (float*)   alloc((size_t)N * 4 * 4);
    unsigned* mx1   = (unsigned*)alloc((size_t)N * 4 * 4);
    float*    den1  = (float*)   alloc((size_t)N * 4 * 4);
    float*    ex1   = (float*)   alloc((size_t)E * 4 * 4);
    float*    out1  = (float*)   alloc((size_t)N * 40 * 4);
    float*    as2b  = (float*)   alloc((size_t)N * 4);
    float*    ad2b  = (float*)   alloc((size_t)N * 4);
    unsigned* mx2   = (unsigned*)alloc((size_t)N * 4);
    float*    den2  = (float*)   alloc((size_t)N * 4);
    float*    ex2   = (float*)   alloc((size_t)E * 4);

    const int B = 256;
    auto nb = [&](long long total) { return (int)((total + B - 1) / B); };

    // weight conversion + transpose (to Ncols x K, f16, zero-padded)
    k_transpose_pad<<<nb(256 * 128), B, 0, stream>>>(W1, W1t, 128, 256, 128, 256);
    k_transpose_pad<<<nb(256 * 256), B, 0, stream>>>(W2, W2t, 256, 256, 256, 256);
    k_transpose_pad<<<nb(128 * 256), B, 0, stream>>>(W3, W3t, 256, 16, 256, 128);
    k_transpose_pad<<<nb(128 * 64),  B, 0, stream>>>(Wg2, Wg2t, 40, 128, 64, 128);

    // x -> f16, rows padded to Mpad
    k_f32_to_f16_pad_rows<<<nb((long long)Mpad * 128), B, 0, stream>>>(
        x, bufA, N, 128, Mpad * 128);

    // MLP: L1 (relu, f16 out), L2 (relu, f16 out), L3 (f32 out, cols padded to 128)
    dim3 gB(Mpad / TM, 2);
    gemm_f16_wmma<<<gB, B, 0, stream>>>(bufA, W1t, b1, nullptr, bufB, 128, 256, 1);
    gemm_f16_wmma<<<gB, B, 0, stream>>>(bufB, W2t, b2, nullptr, bufA, 256, 256, 1);
    dim3 g1(Mpad / TM, 1);
    gemm_f16_wmma<<<g1, B, 0, stream>>>(bufA, W3t, nullptr, fbuf, nullptr, 256, 128, 0);

    // ---- GATConv1 (heads=4, C=10) ----
    k_gat1_lin<<<nb((long long)N * 40), B, 0, stream>>>(fbuf, b3, Wg1, hg1, N);
    k_att_node<<<nb((long long)N * 4), B, 0, stream>>>(hg1, 10, 4, as1, ad1, as1b, ad1b, N);
    hipMemsetAsync(mx1, 0, (size_t)N * 4 * 4, stream);
    hipMemsetAsync(den1, 0, (size_t)N * 4 * 4, stream);
    hipMemsetAsync(out1, 0, (size_t)N * 40 * 4, stream);
    k_edge_max<<<nb((long long)E * 4), B, 0, stream>>>(sE, tE, as1b, ad1b, mx1, E, 4);
    k_edge_expsum<<<nb((long long)E * 4), B, 0, stream>>>(sE, tE, as1b, ad1b, mx1, ex1, den1, E, 4);
    k_aggregate1<<<nb((long long)E * 4), B, 0, stream>>>(sE, tE, ex1, den1, hg1, out1, E);
    k_relu_bias_to_f16pad<<<nb((long long)Mpad * 64), B, 0, stream>>>(
        out1, bg1, h1g16, N, Mpad * 64);

    // ---- GATConv2 (heads=1, C=128): lin via WMMA (K padded 40->64) ----
    gemm_f16_wmma<<<g1, B, 0, stream>>>(h1g16, Wg2t, nullptr, fbuf, nullptr, 64, 128, 0);
    k_att_node<<<nb((long long)N), B, 0, stream>>>(fbuf, 128, 1, as2, ad2, as2b, ad2b, N);
    hipMemsetAsync(mx2, 0, (size_t)N * 4, stream);
    hipMemsetAsync(den2, 0, (size_t)N * 4, stream);
    hipMemsetAsync(out, 0, (size_t)N * 128 * 4, stream);
    k_edge_max<<<nb((long long)E), B, 0, stream>>>(sE, tE, as2b, ad2b, mx2, E, 1);
    k_edge_expsum<<<nb((long long)E), B, 0, stream>>>(sE, tE, as2b, ad2b, mx2, ex2, den2, E, 1);
    k_aggregate2<<<nb((long long)E * 128), B, 0, stream>>>(sE, tE, ex2, den2, fbuf, out, E * 128);
    k_final_sigmoid<<<nb((long long)N * 128), B, 0, stream>>>(out, bg2, N * 128);
}
